// MSA_38044820308629
// MI455X (gfx1250) — matrix operand order repaired and checked by reading, not compiled, not run on hardware
//
#include <hip/hip_runtime.h>

// L2-distance multi-head attention for MI455X (gfx1250, wave32, WMMA).
// v4: one-time f16 conversion of inputs/weights; block-tiled 64x64 GEMMs with
// double-buffered async-to-LDS staging (b128, ASYNCcnt pipelined); 4-wave
// streaming-softmax attention with async-staged K/V tiles.

typedef __attribute__((ext_vector_type(16))) _Float16 v16h;
typedef __attribute__((ext_vector_type(4)))  _Float16 v4h;
typedef __attribute__((ext_vector_type(8)))  float    v8f;
typedef __attribute__((ext_vector_type(4)))  int      v4i;

#define D_MODEL 768
#define NUM_HEADS 12
#define DEPTH 64
#define BATCH 4
#define SEQ 1024
#define ROWS (BATCH * SEQ)  // 4096

#if defined(__AMDGCN__) && __has_builtin(__builtin_amdgcn_global_load_async_to_lds_b128) && \
    __has_builtin(__builtin_amdgcn_s_wait_asynccnt)
#define HAVE_ASYNC 1
#else
#define HAVE_ASYNC 0
#endif

// ---------------------------------------------------------------------------
// Async b128 global->LDS copy (gfx1250 path), with a synchronous fallback.
// ---------------------------------------------------------------------------
__device__ __forceinline__ void async_copy_b128(const _Float16* gsrc,
                                                _Float16* ldst) {
#if HAVE_ASYNC
  typedef __attribute__((address_space(1))) v4i gv4i;
  typedef __attribute__((address_space(3))) v4i lv4i;
  __builtin_amdgcn_global_load_async_to_lds_b128((gv4i*)gsrc, (lv4i*)ldst, 0, 0);
#else
  *(float4*)ldst = *(const float4*)gsrc;
#endif
}

template <int N>
__device__ __forceinline__ void async_wait_le() {
#if HAVE_ASYNC
  __builtin_amdgcn_s_wait_asynccnt(N);
#endif
}

// ---------------------------------------------------------------------------
// WMMA fragment helpers (A and B share contiguous k = 16*half + s order, wave32)
// ---------------------------------------------------------------------------

__device__ __forceinline__ int a_frag_k(int p, int half) {
  // 16-bit A 16x32: VGPR p -> K = half*16 + 2p(+1) (matches B fragment k order)
  return (half << 4) + (p << 1);
}

// A-matrix 16x32 fragment; src points at tile element (0,0), row stride ld.
__device__ __forceinline__ v16h load_a_frag(const _Float16* src, int ld) {
  const int lane = threadIdx.x & 31;
  const int m = lane & 15, half = lane >> 4;
  v16h a;
#pragma unroll
  for (int p = 0; p < 8; ++p) {
    const int k = a_frag_k(p, half);
    a[2 * p]     = src[m * ld + k];
    a[2 * p + 1] = src[m * ld + k + 1];
  }
  return a;
}

// B-matrix 32x16 fragment, row index = K; element (k,n) = src[k*ld + n].
__device__ __forceinline__ v16h load_b_frag(const _Float16* src, int ld) {
  const int lane = threadIdx.x & 31;
  const int n = lane & 15, kg = (lane >> 4) << 4;
  v16h b;
#pragma unroll
  for (int p = 0; p < 8; ++p) {
    const int k = kg + 2 * p;
    b[2 * p]     = src[k * ld + n];
    b[2 * p + 1] = src[(k + 1) * ld + n];
  }
  return b;
}

// B-matrix 32x16 fragment from a transposed source: element (k,n) = src[n*ld + k].
__device__ __forceinline__ v16h load_bT_frag(const _Float16* src, int ld) {
  const int lane = threadIdx.x & 31;
  const int n = lane & 15, kg = (lane >> 4) << 4;
  v16h b;
#pragma unroll
  for (int p = 0; p < 8; ++p) {
    const int k = kg + 2 * p;
    b[2 * p]     = src[n * ld + k];
    b[2 * p + 1] = src[n * ld + k + 1];
  }
  return b;
}

__device__ __forceinline__ v8f wmma_f16(v16h a, v16h b, v8f c) {
  return __builtin_amdgcn_wmma_f32_16x16x32_f16(false, a, false, b, (short)0, c,
                                                false, false);
}

// ---------------------------------------------------------------------------
// Kernel 0: vectorized f32 -> f16 conversion (inputs and weights, once)
// ---------------------------------------------------------------------------
__global__ void cvt_f32_to_f16(const float* __restrict__ src,
                               _Float16* __restrict__ dst, int n4) {
  const int i = blockIdx.x * blockDim.x + threadIdx.x;
  if (i >= n4) return;
  const float4 x = ((const float4*)src)[i];
  v4h y;
  y[0] = (_Float16)x.x;
  y[1] = (_Float16)x.y;
  y[2] = (_Float16)x.z;
  y[3] = (_Float16)x.w;
  ((v4h*)dst)[i] = y;
}

// ---------------------------------------------------------------------------
// Staging helpers for the tiled GEMM (128-thread block, all async b128).
// ---------------------------------------------------------------------------
// X tile: 64 rows x 32 halves (4 KB), global row stride D_MODEL.
__device__ __forceinline__ void stage_x64x32(const _Float16* g, _Float16* l) {
  const int t = threadIdx.x;
#pragma unroll
  for (int i = 0; i < 2; ++i) {
    const int c = t + i * 128;               // 256 x 16B chunks
    const int row = c >> 2, off = (c & 3) * 8;
    async_copy_b128(g + (size_t)row * D_MODEL + off, l + row * 32 + off);
  }
}
// W tile: 32 rows x 64 halves (4 KB), global row stride D_MODEL.
__device__ __forceinline__ void stage_w32x64(const _Float16* g, _Float16* l) {
  const int t = threadIdx.x;
#pragma unroll
  for (int i = 0; i < 2; ++i) {
    const int c = t + i * 128;
    const int row = c >> 3, off = (c & 7) * 8;
    async_copy_b128(g + (size_t)row * D_MODEL + off, l + row * 64 + off);
  }
}
// 32x64 f16 tile (4 KB), contiguous rows of DEPTH halves.
__device__ __forceinline__ void stage_tile_4k(const _Float16* g, _Float16* l) {
  const int t = threadIdx.x;
  async_copy_b128(g + t * 8, l + t * 8);
  async_copy_b128(g + 1024 + t * 8, l + 1024 + t * 8);
}

// ---------------------------------------------------------------------------
// Kernel 1: tiled GEMM  Y = X @ W  (f16 x f16 -> f32 acc).
// Block (4 waves) computes a 64x64 tile; X/W tiles double-buffered in LDS.
// HEADSPLIT: store f16 in [B,H,S,DEPTH]; else store f32 in [ROWS,D_MODEL].
// ---------------------------------------------------------------------------
template <bool HEADSPLIT>
__global__ void __launch_bounds__(128)
gemm64x64(const _Float16* __restrict__ X, const _Float16* __restrict__ W,
          _Float16* __restrict__ Yh, float* __restrict__ Yf) {
  __shared__ _Float16 xbuf[2][64 * 32];  // 2 x 4 KB
  __shared__ _Float16 wbuf[2][32 * 64];  // 2 x 4 KB

  const int wave = threadIdx.x >> 5;
  const int lane = threadIdx.x & 31;
  const int CB = D_MODEL / 64;  // 12 column blocks
  const int row0 = (blockIdx.x / CB) * 64;
  const int col0 = (blockIdx.x % CB) * 64;

  // Prime the pipeline.
  stage_x64x32(X + (size_t)row0 * D_MODEL, xbuf[0]);
  stage_w32x64(W + col0, wbuf[0]);

  v8f acc[4] = {};
  for (int kb = 0; kb < D_MODEL; kb += 32) {
    const int cur = (kb >> 5) & 1;
    if (kb + 32 < D_MODEL) {
      stage_x64x32(X + (size_t)row0 * D_MODEL + kb + 32, xbuf[cur ^ 1]);
      stage_w32x64(W + (size_t)(kb + 32) * D_MODEL + col0, wbuf[cur ^ 1]);
      async_wait_le<4>();  // the 4 older ops (current tile) have retired
    } else {
      async_wait_le<0>();
    }
    __syncthreads();

    const v16h a = load_a_frag(&xbuf[cur][(wave * 16) * 32], 32);
#pragma unroll
    for (int c = 0; c < 4; ++c) {
      const v16h b = load_b_frag(&wbuf[cur][c * 16], 64);
      acc[c] = wmma_f16(a, b, acc[c]);
    }
    __syncthreads();
  }

  const int n = lane & 15, half = lane >> 4;
  if constexpr (HEADSPLIT) {
    const int h = col0 >> 6;  // tile spans exactly one head
#pragma unroll
    for (int c = 0; c < 4; ++c) {
#pragma unroll
      for (int r = 0; r < 8; ++r) {
        const int row = row0 + wave * 16 + r + (half << 3);
        const int bi = row >> 10, s = row & 1023;
        Yh[((size_t)(bi * NUM_HEADS + h) * SEQ + s) * DEPTH + c * 16 + n] =
            (_Float16)acc[c][r];
      }
    }
  } else {
#pragma unroll
    for (int c = 0; c < 4; ++c) {
#pragma unroll
      for (int r = 0; r < 8; ++r) {
        const int row = row0 + wave * 16 + r + (half << 3);
        Yf[(size_t)row * D_MODEL + col0 + c * 16 + n] = acc[c][r];
      }
    }
  }
}

// ---------------------------------------------------------------------------
// Kernel 2: per-row squared norms over the head dim (for the L2 expansion)
// ---------------------------------------------------------------------------
__global__ void row_norm_kernel(const _Float16* __restrict__ Yh,
                                float* __restrict__ norms, int nrows) {
  const int i = blockIdx.x * blockDim.x + threadIdx.x;
  if (i >= nrows) return;
  const _Float16* p = Yh + (size_t)i * DEPTH;
  float s = 0.f;
#pragma unroll
  for (int d = 0; d < DEPTH; ++d) {
    const float x = (float)p[d];
    s += x * x;
  }
  norms[i] = s;
}

// ---------------------------------------------------------------------------
// Kernel 3: streaming-softmax L2-distance attention.
// 4 waves/block share one (b,h); each wave owns 16 query rows (64 per block).
// K/V tiles (32 keys x 64) are double-buffered in LDS via async-to-LDS.
// ---------------------------------------------------------------------------
__global__ void __launch_bounds__(128)
attention_kernel(const _Float16* __restrict__ Qh, const _Float16* __restrict__ Kh,
                 const _Float16* __restrict__ Vh, const float* __restrict__ qsq,
                 const float* __restrict__ ksq, const float* __restrict__ mask,
                 _Float16* __restrict__ ctx) {
  __shared__ _Float16 kbuf[2][32 * DEPTH];  // 2 x 4 KB
  __shared__ _Float16 vbuf[2][32 * DEPTH];  // 2 x 4 KB
  __shared__ _Float16 pbuf[4][16 * 32];     // per-wave P staging, 4 KB

  const int wave = threadIdx.x >> 5;
  const int lane = threadIdx.x & 31;
  const int nlo = lane & 15, half = lane >> 4;

  const int bh = blockIdx.x >> 4;          // 48 (b,h) pairs
  const int qsup = blockIdx.x & 15;        // 16 super-blocks of 64 q rows
  const int b = bh / NUM_HEADS;
  const int h = bh % NUM_HEADS;
  const int row0 = qsup * 64 + wave * 16;  // this wave's 16 query rows

  const _Float16* Q = Qh + (size_t)bh * SEQ * DEPTH;
  const _Float16* K = Kh + (size_t)bh * SEQ * DEPTH;
  const _Float16* V = Vh + (size_t)bh * SEQ * DEPTH;
  const float* qs = qsq + (size_t)bh * SEQ;
  const float* ks = ksq + (size_t)bh * SEQ;

  // Q fragments (A-layout) over feature dims 0..31 / 32..63 — loaded once.
  const v16h aq0 = load_a_frag(&Q[(size_t)row0 * DEPTH], DEPTH);
  const v16h aq1 = load_a_frag(&Q[(size_t)row0 * DEPTH + 32], DEPTH);

  float qsr[8];
#pragma unroll
  for (int r = 0; r < 8; ++r) qsr[r] = qs[row0 + r + (half << 3)];

  float m_run[8], l_run[8];
#pragma unroll
  for (int r = 0; r < 8; ++r) {
    m_run[r] = -3.0e38f;
    l_run[r] = 0.f;
  }
  v8f Oacc[4] = {};

  // Prime the pipeline: stage tile 0 into buffer 0.
  stage_tile_4k(&K[0], kbuf[0]);
  stage_tile_4k(&V[0], vbuf[0]);

  for (int kb = 0; kb < SEQ; kb += 32) {
    const int cur = (kb >> 5) & 1;
    if (kb + 32 < SEQ) {
      stage_tile_4k(&K[(size_t)(kb + 32) * DEPTH], kbuf[cur ^ 1]);
      stage_tile_4k(&V[(size_t)(kb + 32) * DEPTH], vbuf[cur ^ 1]);
      async_wait_le<4>();
    } else {
      async_wait_le<0>();
    }
    __syncthreads();

    const _Float16* kl = kbuf[cur];
    const _Float16* vl = vbuf[cur];

    // ---- score tiles: s0 = keys [kb,kb+16), s1 = keys [kb+16,kb+32)
    v8f s0 = {}, s1 = {};
    {
      const v16h b00 = load_bT_frag(kl, DEPTH);       // feats 0..31
      const v16h b01 = load_bT_frag(kl + 32, DEPTH);  // feats 32..63
      s0 = wmma_f16(aq0, b00, s0);
      s0 = wmma_f16(aq1, b01, s0);
      const v16h b10 = load_bT_frag(kl + 16 * DEPTH, DEPTH);
      const v16h b11 = load_bT_frag(kl + 16 * DEPTH + 32, DEPTH);
      s1 = wmma_f16(aq0, b10, s1);
      s1 = wmma_f16(aq1, b11, s1);
    }

    const int c0 = kb + nlo, c1 = kb + 16 + nlo;
    const float ks0 = ks[c0], ks1 = ks[c1];
    _Float16* pw = pbuf[wave];

#pragma unroll
    for (int r = 0; r < 8; ++r) {
      const int m = r + (half << 3);
      const int row = row0 + m;
      // L2 expansion: sqrt(max(||q||^2 + ||k||^2 - 2 q.k, 0)) / sqrt(64)
      const float d0 = fmaxf(qsr[r] + ks0 - 2.f * s0[r], 0.f);
      const float d1 = fmaxf(qsr[r] + ks1 - 2.f * s1[r], 0.f);
      float sc0 = sqrtf(d0) * 0.125f + mask[(size_t)row * SEQ + c0] * -1e9f;
      float sc1 = sqrtf(d1) * 0.125f + mask[(size_t)row * SEQ + c1] * -1e9f;

      // row max across the 16 lanes of this half-group
      float mv = fmaxf(sc0, sc1);
#pragma unroll
      for (int off = 1; off < 16; off <<= 1)
        mv = fmaxf(mv, __shfl_xor(mv, off, 32));
      const float newm = fmaxf(m_run[r], mv);

      const float p0 = __expf(sc0 - newm);
      const float p1 = __expf(sc1 - newm);
      float rs = p0 + p1;
#pragma unroll
      for (int off = 1; off < 16; off <<= 1) rs += __shfl_xor(rs, off, 32);

      const float scale = __expf(m_run[r] - newm);
      m_run[r] = newm;
      l_run[r] = l_run[r] * scale + rs;
#pragma unroll
      for (int c = 0; c < 4; ++c) Oacc[c][r] *= scale;

      pw[m * 32 + nlo] = (_Float16)p0;
      pw[m * 32 + 16 + nlo] = (_Float16)p1;
    }

    // Re-read P as an A-layout 16x32 f16 fragment (same-wave LDS, in-order)
    v16h ap;
#pragma unroll
    for (int p = 0; p < 8; ++p) {
      const int k = a_frag_k(p, half);
      ap[2 * p]     = pw[nlo * 32 + k];
      ap[2 * p + 1] = pw[nlo * 32 + k + 1];
    }

    // O += P @ V over the 64-wide head in four 16-col chunks
#pragma unroll
    for (int c = 0; c < 4; ++c) {
      const v16h bv = load_b_frag(vl + c * 16, DEPTH);
      Oacc[c] = wmma_f16(ap, bv, Oacc[c]);
    }
    __syncthreads();  // all waves done with this K/V buffer before overwrite
  }

  // Epilogue: normalize, write context in [B,S,d_model] (merged heads), f16
#pragma unroll
  for (int c = 0; c < 4; ++c) {
#pragma unroll
    for (int r = 0; r < 8; ++r) {
      const int row = row0 + r + (half << 3);
      const float val = Oacc[c][r] / l_run[r];
      ctx[((size_t)(b * SEQ + row)) * D_MODEL + h * DEPTH + c * 16 + nlo] =
          (_Float16)val;
    }
  }
}

// ---------------------------------------------------------------------------
// Launch
// ---------------------------------------------------------------------------
extern "C" void kernel_launch(void* const* d_in, const int* in_sizes, int n_in,
                              void* d_out, int out_size, void* d_ws,
                              size_t ws_size, hipStream_t stream) {
  const float* v    = (const float*)d_in[0];
  const float* k    = (const float*)d_in[1];
  const float* q    = (const float*)d_in[2];
  const float* mask = (const float*)d_in[3];
  const float* wq   = (const float*)d_in[4];
  const float* wk   = (const float*)d_in[5];
  const float* wv   = (const float*)d_in[6];
  const float* wo   = (const float*)d_in[7];
  float* out = (float*)d_out;

  char* ws = (char*)d_ws;
  const size_t mat16 = (size_t)ROWS * D_MODEL * sizeof(_Float16);   // 6 MB
  const size_t wmat16 = (size_t)D_MODEL * D_MODEL * sizeof(_Float16);
  size_t o = 0;
  _Float16* qf  = (_Float16*)(ws + o); o += mat16;   // f16 inputs
  _Float16* kf  = (_Float16*)(ws + o); o += mat16;
  _Float16* vf  = (_Float16*)(ws + o); o += mat16;
  _Float16* wqh = (_Float16*)(ws + o); o += wmat16;  // f16 weights
  _Float16* wkh = (_Float16*)(ws + o); o += wmat16;
  _Float16* wvh = (_Float16*)(ws + o); o += wmat16;
  _Float16* woh = (_Float16*)(ws + o); o += wmat16;
  _Float16* qh  = (_Float16*)(ws + o); o += mat16;   // head-split projections
  _Float16* kh  = (_Float16*)(ws + o); o += mat16;
  _Float16* vh  = (_Float16*)(ws + o); o += mat16;
  _Float16* ctx = (_Float16*)(ws + o); o += mat16;
  float* qsq = (float*)(ws + o); o += (size_t)BATCH * NUM_HEADS * SEQ * 4;
  float* ksq = (float*)(ws + o);

  // One-time f16 conversions.
  const int in4 = ROWS * D_MODEL / 4;       // 786432
  const int w4  = D_MODEL * D_MODEL / 4;    // 147456
  cvt_f32_to_f16<<<(in4 + 255) / 256, 256, 0, stream>>>(q, qf, in4);
  cvt_f32_to_f16<<<(in4 + 255) / 256, 256, 0, stream>>>(k, kf, in4);
  cvt_f32_to_f16<<<(in4 + 255) / 256, 256, 0, stream>>>(v, vf, in4);
  cvt_f32_to_f16<<<(w4 + 255) / 256, 256, 0, stream>>>(wq, wqh, w4);
  cvt_f32_to_f16<<<(w4 + 255) / 256, 256, 0, stream>>>(wk, wkh, w4);
  cvt_f32_to_f16<<<(w4 + 255) / 256, 256, 0, stream>>>(wv, wvh, w4);
  cvt_f32_to_f16<<<(w4 + 255) / 256, 256, 0, stream>>>(wo, woh, w4);

  // Projections (head-split f16 out).
  const dim3 gg((ROWS / 64) * (D_MODEL / 64)), gb(128);  // 768 blocks
  gemm64x64<true><<<gg, gb, 0, stream>>>(qf, wqh, qh, nullptr);
  gemm64x64<true><<<gg, gb, 0, stream>>>(kf, wkh, kh, nullptr);
  gemm64x64<true><<<gg, gb, 0, stream>>>(vf, wvh, vh, nullptr);

  const int nrows = BATCH * NUM_HEADS * SEQ;
  row_norm_kernel<<<(nrows + 255) / 256, 256, 0, stream>>>(qh, qsq, nrows);
  row_norm_kernel<<<(nrows + 255) / 256, 256, 0, stream>>>(kh, ksq, nrows);

  attention_kernel<<<BATCH * NUM_HEADS * (SEQ / 64), 128, 0, stream>>>(
      qh, kh, vh, qsq, ksq, mask, ctx);

  // Output projection (f32 out).
  gemm64x64<false><<<gg, gb, 0, stream>>>(ctx, woh, nullptr, out);
}
